// AnyNet_74474732913041
// MI455X (gfx1250) — compile-verified, matrix operand order checked by reading, not run on hardware
//
#include <hip/hip_runtime.h>
#include <hip/hip_bf16.h>

// AnyNet cost-volume + classif head for MI455X (gfx1250).
// Pipeline (f16 data path, f32 WMMA accumulate):
//   prep_kernel:          fold BN into conv1 weights -> f16 [tap][co][ci]; w2 -> f16
//   halo_zero_kernel:     zero the 1-element D/H/W halo of the padded volume
//   gwc_volume_kernel:    groupwise correlation -> f16 volume, channel-last,
//                         halo-padded [b][50][74][242][32]
//   conv1_wmma_kernel:    implicit-GEMM 3^3 conv 32->32 via v_wmma_f32_16x16x32_f16;
//                         weights staged to LDS by the Tensor Data Mover
//                         (tensor_load_to_lds + s_wait_tensorcnt), A fragments via
//                         ds_load_b128, B via unconditional global_load_b128
//   conv2_softmax_kernel: 3^3 conv 32->1 + softmax over 48 disparities + expectation
//                         (w2 staged with global_load_async_to_lds_b128 / ASYNCcnt)

typedef _Float16 v16h __attribute__((ext_vector_type(16)));
typedef float    v8f  __attribute__((ext_vector_type(8)));
typedef _Float16 v2h  __attribute__((ext_vector_type(2)));
typedef unsigned int u32x4 __attribute__((ext_vector_type(4)));
typedef unsigned int u32x8 __attribute__((ext_vector_type(8)));

struct __attribute__((aligned(16))) U4 { unsigned int x, y, z, w; };
union AFrag { v16h h; U4 u[2]; };

#define NB   4
#define CIN  320
#define NH   72
#define NW   240
#define NG   32
#define CPG  10
#define ND   48
// halo-padded volume dims
#define D2   50
#define H2   74
#define W2   242

// ---------------------------------------------------------------------------
// CDNA5 async global->LDS copy (ASYNCcnt path). LDS operand is the low 32
// bits of the generic pointer.
// ---------------------------------------------------------------------------
__device__ __forceinline__ void async_g2l_b128(const void* gsrc, void* ldst) {
    unsigned laddr = (unsigned)(unsigned long long)ldst;
    unsigned long long gaddr = (unsigned long long)gsrc;
    asm volatile("global_load_async_to_lds_b128 %0, %1, off"
                 :: "v"(laddr), "v"(gaddr) : "memory");
}
__device__ __forceinline__ void wait_async0() {
    asm volatile("s_wait_asynccnt 0x0" ::: "memory");
}

// ---------------------------------------------------------------------------
// CDNA5 Tensor Data Mover: 1-D bulk copy global -> LDS (TENSORcnt path).
// D# built per ISA 8.3/8.4: group0 {count=1, lds_addr, global_addr, type=2},
// group1 {data_size=8B, tensor_dim0=tile_dim0=nbytes/8, stride=nbytes/8}.
// Issue from ONE wave per block (EXEC is ignored; it is a wave-level DMA).
// ---------------------------------------------------------------------------
__device__ __forceinline__ void tdm_load_1d(const void* gsrc, void* ldst,
                                            unsigned nbytes) {
    const unsigned n8 = nbytes >> 3;                  // 8-byte elements
    const unsigned long long ga = (unsigned long long)gsrc;
    u32x4 g0;
    g0[0] = 1u;                                       // count=1, user mode
    g0[1] = (unsigned)(unsigned long long)ldst;       // lds_addr
    g0[2] = (unsigned)ga;                             // global_addr[31:0]
    g0[3] = (unsigned)((ga >> 32) & 0x01FFFFFFu) | (2u << 30);  // addr[56:32]|type=2
    u32x8 g1;
    g1[0] = 3u << 16;                                 // data_size = 8 bytes
    g1[1] = (n8 & 0xFFFFu) << 16;                     // tensor_dim0[15:0]
    g1[2] = (n8 >> 16);                               // tensor_dim0[31:16], dim1=0
    g1[3] = (n8 & 0xFFFFu) << 16;                     // tile_dim0
    g1[4] = 0u;                                       // tile_dim1/2 unused
    g1[5] = n8;                                       // tensor_dim0_stride lo
    g1[6] = 0u;
    g1[7] = 0u;
    asm volatile("tensor_load_to_lds %0, %1" :: "s"(g0), "s"(g1) : "memory");
}
__device__ __forceinline__ void wait_tensor0() {
    __builtin_amdgcn_s_wait_tensorcnt(0);
}

// ---------------------------------------------------------------------------
// Kernel 0: weight prep. Fold BN (eval) scale into conv1 weights, cast f16,
// repack OIDHW -> [tap][co][ci] (ci contiguous = WMMA A-fragment friendly).
// ---------------------------------------------------------------------------
__global__ __launch_bounds__(256)
void prep_kernel(const float* __restrict__ w1, const float* __restrict__ gamma,
                 const float* __restrict__ beta, const float* __restrict__ mean,
                 const float* __restrict__ var, const float* __restrict__ w2,
                 _Float16* __restrict__ wf, float* __restrict__ bias,
                 _Float16* __restrict__ w2f) {
    const int tid = threadIdx.x;
    for (int idx = tid; idx < 27 * 32 * 32; idx += 256) {
        int t  = idx >> 10;
        int co = (idx >> 5) & 31;
        int ci = idx & 31;
        int dd = t / 9, r = t % 9, dh = r / 3, dw = r % 3;
        float s = gamma[co] * rsqrtf(var[co] + 1e-5f);
        wf[idx] = (_Float16)(w1[(((co * 32 + ci) * 3 + dd) * 3 + dh) * 3 + dw] * s);
    }
    for (int idx = tid; idx < 27 * 32; idx += 256) {
        int t = idx >> 5, ci = idx & 31;
        int dd = t / 9, r = t % 9, dh = r / 3, dw = r % 3;
        w2f[idx] = (_Float16)w2[((ci * 3 + dd) * 3 + dh) * 3 + dw];
    }
    if (tid < 32) {
        float s = gamma[tid] * rsqrtf(var[tid] + 1e-5f);
        bias[tid] = beta[tid] - mean[tid] * s;
    }
}

// ---------------------------------------------------------------------------
// Kernel 0b: zero the halo (d2 in {0,49} | h2 in {0,73} | w2 in {0,241}).
// 263,840 border positions x 64B -> ~17 MB of stores.
// ---------------------------------------------------------------------------
#define NFA (NB * 2 * H2 * W2)        // d faces
#define NFB (NB * ND * 2 * W2)        // h faces (d interior)
#define NFC (NB * ND * NH * 2)        // w faces (d,h interior)
__global__ __launch_bounds__(256)
void halo_zero_kernel(_Float16* __restrict__ vol) {
    const int idx = blockIdx.x * 256 + threadIdx.x;
    if (idx >= NFA + NFB + NFC) return;
    int b, d2, h2, w2;
    if (idx < NFA) {
        b = idx / (2 * H2 * W2);
        int r = idx % (2 * H2 * W2);
        d2 = (r / (H2 * W2)) ? (D2 - 1) : 0;
        int r2 = r % (H2 * W2);
        h2 = r2 / W2; w2 = r2 % W2;
    } else if (idx < NFA + NFB) {
        int j = idx - NFA;
        b = j / (ND * 2 * W2);
        int r = j % (ND * 2 * W2);
        d2 = 1 + r / (2 * W2);
        int r2 = r % (2 * W2);
        h2 = (r2 / W2) ? (H2 - 1) : 0;
        w2 = r2 % W2;
    } else {
        int j = idx - NFA - NFB;
        b = j / (ND * NH * 2);
        int r = j % (ND * NH * 2);
        d2 = 1 + r / (NH * 2);
        int r2 = r % (NH * 2);
        h2 = 1 + r2 / 2;
        w2 = (r2 & 1) ? (W2 - 1) : 0;
    }
    U4 z; z.x = z.y = z.z = z.w = 0u;
    U4* p = (U4*)(vol + ((((size_t)b * D2 + d2) * H2 + h2) * W2 + w2) * NG);
    p[0] = z; p[1] = z; p[2] = z; p[3] = z;
}

// ---------------------------------------------------------------------------
// Kernel 1: group-wise correlation volume, channel-last f16, halo-padded out.
// Block = (w-tile of 16, h, b). LDS holds the 320-channel left row tile and
// the shifted right row range [w0-48, w0+15] as f16 ([w][c] layout so the
// 10-channel group dot is contiguous -> v2h packed LDS loads).
// ---------------------------------------------------------------------------
__global__ __launch_bounds__(256)
void gwc_volume_kernel(const float* __restrict__ fl, const float* __restrict__ fr,
                       _Float16* __restrict__ vol) {
    __shared__ __attribute__((aligned(16))) _Float16 sF[16 * CIN];   // 10 KB
    __shared__ __attribute__((aligned(16))) _Float16 sR[64 * CIN];   // 40 KB
    const int w0 = blockIdx.x * 16;
    const int h  = blockIdx.y;
    const int b  = blockIdx.z;
    const int tid = threadIdx.x;

    for (int idx = tid; idx < 16 * CIN; idx += 256) {
        int c = idx >> 4, w = idx & 15;
        sF[w * CIN + c] = (_Float16)fl[(((size_t)b * CIN + c) * NH + h) * NW + w0 + w];
    }
    for (int idx = tid; idx < 64 * CIN; idx += 256) {
        int c = idx >> 6, w = idx & 63;
        int gw = w0 - 48 + w;                       // global w of right feature
        float v = (gw >= 0) ? fr[(((size_t)b * CIN + c) * NH + h) * NW + gw] : 0.0f;
        sR[w * CIN + c] = (_Float16)v;
    }
    __syncthreads();

    const int g  = tid & 31;
    const int ws = tid >> 5;        // this thread covers w = ws and ws+8
    const int cbase = g * CPG;

    float flv[2][CPG];
    for (int k = 0; k < 2; ++k) {
        const v2h* p = (const v2h*)&sF[(ws + 8 * k) * CIN + cbase];
        for (int i = 0; i < 5; ++i) {
            v2h t = p[i];
            flv[k][2 * i] = (float)t[0];
            flv[k][2 * i + 1] = (float)t[1];
        }
    }
    for (int d = 0; d < ND; ++d) {
        for (int k = 0; k < 2; ++k) {
            int w = ws + 8 * k;
            const v2h* p = (const v2h*)&sR[(w - d + 48) * CIN + cbase];
            float acc = 0.0f;
            for (int i = 0; i < 5; ++i) {
                v2h t = p[i];
                acc += flv[k][2 * i] * (float)t[0] + flv[k][2 * i + 1] * (float)t[1];
            }
            // left zero-pad (w < d) falls out automatically: sR holds 0 there
            vol[((((size_t)b * D2 + (d + 1)) * H2 + (h + 1)) * W2 + (w0 + w + 1)) * NG + g] =
                (_Float16)(acc * 0.1f);
        }
    }
}

// ---------------------------------------------------------------------------
// Kernel 2: conv1 (3x3x3, 32->32) as implicit GEMM with WMMA f16.
// One wave per 16-wide W tile at fixed (b,d,h). Two co M-tiles, 27 taps of
// K=32 -> 54 v_wmma_f32_16x16x32_f16 per wave, fully unrolled so the
// scheduler can cover the f16 WMMA->VALU hazard slots with useful work.
// Weights staged once per block by the Tensor Data Mover; A fragments via
// ds_load_b128; B fragments via unconditional global_load_b128 (halo pad).
// Fragment addressing follows the ISA 16-bit layouts (lanes 0-15: K 0-7 &
// 16-23; lanes 16-31: K 8-15 & 24-31).
// ---------------------------------------------------------------------------
__global__ __launch_bounds__(256)
void conv1_wmma_kernel(const _Float16* __restrict__ vol, const _Float16* __restrict__ wf,
                       const float* __restrict__ bias, _Float16* __restrict__ act) {
    __shared__ __attribute__((aligned(16))) _Float16 sW[27 * 32 * 32];  // 54 KB
    const int tid  = threadIdx.x;
    const int lane = tid & 31;
    const int wave = tid >> 5;

    // TDM: one wave-level DMA brings all folded weights into LDS.
    if (wave == 0) {
        if (lane == 0) tdm_load_1d(wf, sW, 27 * 32 * 32 * sizeof(_Float16));
        wait_tensor0();
    }
    __syncthreads();

    int id = blockIdx.x * 8 + wave;
    const int tw = id % 15; id /= 15;
    const int h  = id % NH; id /= NH;
    const int d  = id % ND; id /= ND;
    const int b  = id;
    const int w0 = tw * 16;

    const int m   = lane & 15;      // A row (co within tile) / B column
    const int hg  = lane >> 4;      // lane half-group
    const int ci0 = hg * 8;         // first K element held by this lane

    v8f c0 = {0.f, 0.f, 0.f, 0.f, 0.f, 0.f, 0.f, 0.f};
    v8f c1 = c0;

#pragma unroll
    for (int dd = 0; dd < 3; ++dd) {
#pragma unroll
        for (int dh = 0; dh < 3; ++dh) {
            // padded plane: dp+1 = d+dd, hp+1 = h+dh (always in range)
            const _Float16* rowBase =
                vol + ((((size_t)b * D2 + (d + dd)) * H2 + (h + dh)) * W2) * NG;
            __builtin_prefetch(rowBase + (size_t)(w0 + m + 1) * NG, 0, 3);
#pragma unroll
            for (int dw = 0; dw < 3; ++dw) {
                const int tap = (dd * 3 + dh) * 3 + dw;
                AFrag a0, a1, bf;
                {   // A fragments from LDS: [tap][co][ci], ci contiguous
                    const U4* p = (const U4*)(sW + ((size_t)(tap * 32 + m) * 32 + ci0));
                    a0.u[0] = p[0]; a0.u[1] = p[2];          // K ci0..+7, ci0+16..+23
                    const U4* q = (const U4*)(sW + ((size_t)(tap * 32 + 16 + m) * 32 + ci0));
                    a1.u[0] = q[0]; a1.u[1] = q[2];
                }
                {   // B fragment: padded column index w0+m+dw, never OOB
                    const U4* p = (const U4*)(rowBase + (size_t)(w0 + m + dw) * NG + ci0);
                    bf.u[0] = p[0]; bf.u[1] = p[2];
                }
                c0 = __builtin_amdgcn_wmma_f32_16x16x32_f16(false, a0.h, false, bf.h,
                                                            (short)0, c0, false, false);
                c1 = __builtin_amdgcn_wmma_f32_16x16x32_f16(false, a1.h, false, bf.h,
                                                            (short)0, c1, false, false);
            }
        }
    }

    // D layout: VGPR r, lane -> M = r + 8*hg, N = m. Store 8 contiguous f16
    // channels per tile with one b128 each (channel-last activation).
    const int cobase = hg * 8;
    union { U4 u; _Float16 hv[8]; } o0, o1;
    for (int r = 0; r < 8; ++r) {
        o0.hv[r] = (_Float16)fmaxf(c0[r] + bias[cobase + r], 0.0f);
        o1.hv[r] = (_Float16)fmaxf(c1[r] + bias[16 + cobase + r], 0.0f);
    }
    _Float16* outp = act + ((((size_t)b * ND + d) * NH + h) * NW + (w0 + m)) * NG;
    *(U4*)(outp + cobase)      = o0.u;
    *(U4*)(outp + 16 + cobase) = o1.u;
}

// ---------------------------------------------------------------------------
// Kernel 3: conv2 (3x3x3, 32->1) fused with softmax over disparity and the
// disparity expectation. Block = (16-wide w tile, h, b); all 48 d in LDS.
// ---------------------------------------------------------------------------
__global__ __launch_bounds__(256)
void conv2_softmax_kernel(const _Float16* __restrict__ act,
                          const _Float16* __restrict__ w2f,
                          float* __restrict__ out) {
    __shared__ __attribute__((aligned(16))) _Float16 sw[27 * 32];
    __shared__ float zb[ND * 16];
    const int w0 = blockIdx.x * 16;
    const int h  = blockIdx.y;
    const int b  = blockIdx.z;
    const int tid = threadIdx.x;

    if (tid < 27 * 32 / 8)                       // 108 x 16B async chunks
        async_g2l_b128(w2f + (size_t)tid * 8, sw + (size_t)tid * 8);
    wait_async0();
    __syncthreads();

    for (int k = 0; k < 3; ++k) {
        const int idx = tid + k * 256;      // 768 = 48 d x 16 w outputs
        const int d = idx >> 4;
        const int w = idx & 15;
        float acc = 0.0f;
        for (int dd = 0; dd < 3; ++dd) {
            int dp = d + dd - 1; if (dp < 0 || dp >= ND) continue;
            for (int dh = 0; dh < 3; ++dh) {
                int hp = h + dh - 1; if (hp < 0 || hp >= NH) continue;
                for (int dw = 0; dw < 3; ++dw) {
                    int wp = w0 + w + dw - 1; if (wp < 0 || wp >= NW) continue;
                    int tap = (dd * 3 + dh) * 3 + dw;
                    const v2h* pa =
                        (const v2h*)(act + ((((size_t)b * ND + dp) * NH + hp) * NW + wp) * NG);
                    const v2h* pw = (const v2h*)(sw + tap * 32);
                    for (int i = 0; i < 16; ++i) {
                        v2h a = pa[i], ww = pw[i];
                        acc += (float)a[0] * (float)ww[0] + (float)a[1] * (float)ww[1];
                    }
                }
            }
        }
        zb[idx] = acc;
    }
    __syncthreads();

    if (tid < 16) {
        float m = -1e30f;
        for (int d = 0; d < ND; ++d) m = fmaxf(m, zb[d * 16 + tid]);
        float den = 0.0f, num = 0.0f;
        for (int d = 0; d < ND; ++d) {
            float e = __expf(zb[d * 16 + tid] - m);
            den += e;
            num += e * (float)d;
        }
        out[((size_t)b * NH + h) * NW + w0 + tid] = num / den;
    }
}

// ---------------------------------------------------------------------------
extern "C" void kernel_launch(void* const* d_in, const int* in_sizes, int n_in,
                              void* d_out, int out_size, void* d_ws, size_t ws_size,
                              hipStream_t stream) {
    (void)in_sizes; (void)n_in; (void)out_size; (void)ws_size;
    const float* fl    = (const float*)d_in[0];
    const float* fr    = (const float*)d_in[1];
    const float* w1    = (const float*)d_in[2];
    const float* gamma = (const float*)d_in[3];
    const float* beta  = (const float*)d_in[4];
    const float* mean  = (const float*)d_in[5];
    const float* var   = (const float*)d_in[6];
    const float* w2    = (const float*)d_in[7];
    float* out = (float*)d_out;

    char* ws = (char*)d_ws;
    const size_t volPadBytes = (size_t)NB * D2 * H2 * W2 * NG * sizeof(_Float16); // ~229 MB
    const size_t actBytes    = (size_t)NB * ND * NH * NW * NG * sizeof(_Float16); // ~212 MB
    _Float16* vol  = (_Float16*)ws;
    _Float16* act  = (_Float16*)(ws + volPadBytes);
    _Float16* wf   = (_Float16*)(ws + volPadBytes + actBytes);
    float*    bias = (float*)   (ws + volPadBytes + actBytes + 27 * 32 * 32 * sizeof(_Float16));
    _Float16* w2f  = (_Float16*)(ws + volPadBytes + actBytes + 27 * 32 * 32 * sizeof(_Float16) + 256);

    prep_kernel<<<1, 256, 0, stream>>>(w1, gamma, beta, mean, var, w2, wf, bias, w2f);
    halo_zero_kernel<<<dim3((NFA + NFB + NFC + 255) / 256), 256, 0, stream>>>(vol);
    gwc_volume_kernel<<<dim3(NW / 16, NH, NB), 256, 0, stream>>>(fl, fr, vol);
    // one wave (of 8 per block) per 16-wide W tile: 4*48*72*15 tiles / 8
    conv1_wmma_kernel<<<dim3((NB * ND * NH * 15) / 8), 256, 0, stream>>>(vol, wf, bias, act);
    conv2_softmax_kernel<<<dim3(NW / 16, NH, NB), 256, 0, stream>>>(act, w2f, out);
}
